// NnHalfKPCuda_29566554865848
// MI455X (gfx1250) — matrix operand order repaired and checked by reading, not compiled, not run on hardware
//
#include <hip/hip_runtime.h>
#include <hip/hip_bf16.h>

typedef __attribute__((ext_vector_type(16))) _Float16 v16h;
typedef __attribute__((ext_vector_type(8)))  float    v8f;
typedef __attribute__((ext_vector_type(4)))  float    f4;

#define BATCH     8192
#define KACT      32
#define FOUT      512
#define NBUCK     8
#define FFTVOC    640
#define ROWS_PER_BLOCK 16
#define HSTRIDE   1032   // 1024 + 8 halves pad -> conflict-free LDS

// ---------------------------------------------------------------------------
// Prep: pack out_w (8 x 1024 f32, padded to 16 cols with zeros) into f16
// B-operand tiles for v_wmma_f32_16x16x32_f16, in contiguous k order (same as A):
//   B is 32x16 (KxN) per k-block t; lane = (k/16)*16 + n, half h = k%16.
// Bpack layout: [t][lane][h], t=0..31, contiguous 16 halves (32B) per lane.
// ---------------------------------------------------------------------------
__global__ void prep_outw_kernel(const float* __restrict__ out_w,
                                 _Float16* __restrict__ Bpack) {
    int idx = blockIdx.x * 256 + threadIdx.x;       // 32*32*16 = 16384 elems
    if (idx >= 32 * 32 * 16) return;
    int h    = idx & 15;
    int lane = (idx >> 4) & 31;
    int t    = idx >> 9;
    int n    = lane & 15;
    int kg   = 32 * t + (lane >> 4) * 16 + h;       // global k in [0,1024)
    float v  = (n < NBUCK) ? out_w[n * (2 * FOUT) + kg] : 0.0f;
    Bpack[idx] = (_Float16)v;
}

// ---------------------------------------------------------------------------
// Main kernel: 16 waves/block, one batch row per wave.
// ---------------------------------------------------------------------------
__global__ __launch_bounds__(512) void nnue_fwd_kernel(
    const float* __restrict__ values,
    const int*   __restrict__ stm_idx,
    const int*   __restrict__ nstm_idx,
    const int*   __restrict__ buckets,
    const float* __restrict__ ft_w,
    const float* __restrict__ ft_b,
    const float* __restrict__ fft_w,
    const float* __restrict__ fft_b,
    const float* __restrict__ out_b,
    const _Float16* __restrict__ Bpack,
    float* __restrict__ out)
{
    __shared__ _Float16 hid[ROWS_PER_BLOCK][HSTRIDE];  // f16 hidden, padded
    __shared__ float    Cst[8][32];                    // C tile staging

    const int lane = threadIdx.x & 31;
    const int row  = __builtin_amdgcn_readfirstlane(threadIdx.x >> 5);
    const int b    = blockIdx.x * ROWS_PER_BLOCK + row;

    // ---- embedding bags: 16 stm + 16 nstm features per lane ----
    float accS[16], accN[16];
#pragma unroll
    for (int i = 0; i < 16; ++i) { accS[i] = 0.0f; accN[i] = 0.0f; }

    const int l4 = lane * 4;
    const int*   si = stm_idx  + b * KACT;   // wave-uniform -> s_load
    const int*   ni = nstm_idx + b * KACT;
    const float* vv = values   + b * KACT;

#pragma unroll 2
    for (int k = 0; k < KACT; ++k) {
        const int   is = si[k];
        const int   in = ni[k];
        const float v  = vv[k];
        const int   fs = is % FFTVOC;
        const int   fn = in % FFTVOC;
        const f4* rs  = (const f4*)(ft_w  + is * FOUT + l4);
        const f4* rfs = (const f4*)(fft_w + fs * FOUT + l4);
        const f4* rn  = (const f4*)(ft_w  + in * FOUT + l4);
        const f4* rfn = (const f4*)(fft_w + fn * FOUT + l4);
#pragma unroll
        for (int c = 0; c < 4; ++c) {
            f4 a0 = rs[c * 32];
            f4 a1 = rfs[c * 32];
            f4 d0 = rn[c * 32];
            f4 d1 = rfn[c * 32];
#pragma unroll
            for (int j = 0; j < 4; ++j) {
                accS[c * 4 + j] += v * (a0[j] + a1[j]);
                accN[c * 4 + j] += v * (d0[j] + d1[j]);
            }
        }
    }

    // ---- bias + clip + f16 hidden into LDS ----
#pragma unroll
    for (int c = 0; c < 4; ++c) {
        f4 b1 = *(const f4*)(ft_b  + c * 128 + l4);
        f4 b2 = *(const f4*)(fft_b + c * 128 + l4);
#pragma unroll
        for (int j = 0; j < 4; ++j) {
            const float bias = b1[j] + b2[j];
            float hs = fminf(fmaxf(accS[c * 4 + j] + bias, 0.0f), 1.0f);
            float hn = fminf(fmaxf(accN[c * 4 + j] + bias, 0.0f), 1.0f);
            const int f = c * 128 + l4 + j;
            hid[row][f]        = (_Float16)hs;
            hid[row][512 + f]  = (_Float16)hn;
        }
    }
    __syncthreads();

    // ---- output layer: wave 0 computes 16x16 logits with WMMA f16 ----
    if (threadIdx.x < 32) {
        v8f c = {};
        const int m  = lane & 15;      // A-matrix row this lane carries
        const int hi = lane >> 4;      // lane-group -> K sub-block select
#pragma unroll 4
        for (int t = 0; t < 32; ++t) {
            // A 16x32 f16 layout: lane l (M=l%16), half h holds
            //   K = (l/16)*16 + h
            v16h a;
#pragma unroll
            for (int h = 0; h < 16; ++h) {
                const int kk = 32 * t + (hi << 4) + h;
                a[h] = hid[m][kk];
            }
            v16h bb = *(const v16h*)(Bpack + (t * 32 + lane) * 16);
            c = __builtin_amdgcn_wmma_f32_16x16x32_f16(
                    false, a, false, bb, (short)0, c, false, false);
        }
        // C layout: VGPR r, lanes 0-15 -> M=r, lanes 16-31 -> M=8+r, N=lane%16
#pragma unroll
        for (int r = 0; r < 8; ++r) Cst[r][lane] = c[r];
    }
    __syncthreads();

    // ---- bucket select + bias + sigmoid ----
    if (threadIdx.x < ROWS_PER_BLOCK) {
        const int m  = threadIdx.x;
        const int bk = buckets[blockIdx.x * ROWS_PER_BLOCK + m];
        const float l1 = Cst[m & 7][bk + ((m >> 3) << 4)] + out_b[bk];
        out[blockIdx.x * ROWS_PER_BLOCK + m] = 1.0f / (1.0f + __expf(-l1));
    }
}

// ---------------------------------------------------------------------------
extern "C" void kernel_launch(void* const* d_in, const int* in_sizes, int n_in,
                              void* d_out, int out_size, void* d_ws, size_t ws_size,
                              hipStream_t stream) {
    const float* values   = (const float*)d_in[0];
    const int*   stm_idx  = (const int*)d_in[1];
    const int*   nstm_idx = (const int*)d_in[2];
    const int*   buckets  = (const int*)d_in[3];
    const float* ft_w     = (const float*)d_in[4];
    const float* ft_b     = (const float*)d_in[5];
    const float* fft_w    = (const float*)d_in[6];
    const float* fft_b    = (const float*)d_in[7];
    const float* out_w    = (const float*)d_in[8];
    const float* out_b    = (const float*)d_in[9];
    float* out = (float*)d_out;

    _Float16* Bpack = (_Float16*)d_ws;   // 32*32*16*2 = 32 KB

    prep_outw_kernel<<<64, 256, 0, stream>>>(out_w, Bpack);
    nnue_fwd_kernel<<<BATCH / ROWS_PER_BLOCK, 512, 0, stream>>>(
        values, stm_idx, nstm_idx, buckets,
        ft_w, ft_b, fft_w, fft_b, out_b, Bpack, out);
}
